// MultiHeadAttention_20375324852236
// MI455X (gfx1250) — compile-verified, matrix operand order checked by reading, not run on hardware
//
#include <hip/hip_runtime.h>

// ---------------------------------------------------------------------------
// Multi-head attention forward, CDNA5 (gfx1250), bf16 WMMA w/ f32 accumulate.
//   1) x -> bf16; w_qkv/w_out -> bf16 transposed
//   2) GEMM (async-LDS double-buffered, 256x128 tile, 64x64/wave)
//        -> Q[B,H,T,hd], K[B,H,T,hd], Vt[B,H,hd,T]
//   3) flash attention (causal, online softmax, async-LDS K/V staging)
//   4) GEMM -> f32 output
// ---------------------------------------------------------------------------

typedef __bf16 bf16;
typedef __attribute__((ext_vector_type(4)))  __bf16 v4bf;
typedef __attribute__((ext_vector_type(8)))  __bf16 v8bf;
typedef __attribute__((ext_vector_type(16))) __bf16 v16bf;
typedef __attribute__((ext_vector_type(8)))  float  v8f;

union BF16x16 { v16bf v; v8bf h[2]; };

// Fragment loader matching the CDNA5 16-bit A/B WMMA layout (16x32 MxK):
// lanes 0-15 hold row M=lane&15; elems 0..7 = K[hf*8..], elems 8..15 = K[16+hf*8..].
static __device__ __forceinline__ v16bf ld_frag(const bf16* rowp, int hf) {
  BF16x16 u;
  u.h[0] = *(const v8bf*)(rowp + hf * 8);
  u.h[1] = *(const v8bf*)(rowp + hf * 8 + 16);
  return u.v;
}

static __device__ __forceinline__ v8f wmma_bf16(v16bf a, v16bf b, v8f c) {
  return __builtin_amdgcn_wmma_f32_16x16x32_bf16(false, a, false, b,
                                                 (short)0, c, false, false);
}

// --- CDNA5 async global->LDS (ASYNCcnt-tracked), 16 bytes per lane ----------
static __device__ __forceinline__ unsigned lds_off(const void* p) {
  return (unsigned)(uintptr_t)p;   // low 32 bits of generic addr = LDS offset
}
static __device__ __forceinline__ void async_ld_b128(unsigned lds_addr,
                                                     const void* gaddr) {
  asm volatile("global_load_async_to_lds_b128 %0, %1, off"
               :: "v"(lds_addr), "v"((unsigned long long)(uintptr_t)gaddr)
               : "memory");
}
static __device__ __forceinline__ void wait_async0() {
  asm volatile("s_wait_asynccnt 0x0" ::: "memory");
}

// ---------------------------------------------------------------------------
// fp32 -> bf16 (x4)
// ---------------------------------------------------------------------------
__global__ __launch_bounds__(256)
void mha_cvt4(const float4* __restrict__ in, v4bf* __restrict__ outp, int n4) {
  int i = blockIdx.x * 256 + threadIdx.x;
  if (i < n4) {
    float4 v = in[i];
    v4bf o;
    o[0] = (bf16)v.x; o[1] = (bf16)v.y; o[2] = (bf16)v.z; o[3] = (bf16)v.w;
    outp[i] = o;
  }
}

// ---------------------------------------------------------------------------
// fp32 [K][N] -> bf16 [N][K]
// ---------------------------------------------------------------------------
__global__ __launch_bounds__(256)
void mha_transpose_cvt(const float* __restrict__ in, bf16* __restrict__ outp,
                       int K, int N) {
  long i = (long)blockIdx.x * 256 + threadIdx.x;
  if (i < (long)K * N) {
    int k = (int)(i / N), n = (int)(i % N);
    outp[(size_t)n * K + k] = (bf16)in[i];
  }
}

// ---------------------------------------------------------------------------
// Tiled GEMM: C[M,N] = A[M,K] @ Bt[N,K]^T + bias
//   8 waves (4x2), block tile 256x128x32, wave tile 64x64 (16 WMMA / k-step,
//   16 ds_load_b128 / k-step -> 1.0 WMMA per LDS load).
//   Double-buffered LDS fed by global_load_async_to_lds_b128.
//   mode 0: f32 out; mode 1: scatter to Q/K/Vt bf16 head-major.
// ---------------------------------------------------------------------------
__global__ __launch_bounds__(256)
void mha_gemm_wmma(const bf16* __restrict__ A, const bf16* __restrict__ Bt,
                   const float* __restrict__ bias,
                   float* __restrict__ outF,
                   bf16* __restrict__ Qb, bf16* __restrict__ Kb,
                   bf16* __restrict__ Vt,
                   int M, int N, int K, int mode) {
  __shared__ bf16 As[2][256][32];   // 2 x 16 KB
  __shared__ bf16 Bs[2][128][32];   // 2 x  8 KB
  const int tid  = threadIdx.x;
  const int wave = tid >> 5, lane = tid & 31;
  const int wm = wave >> 1, wn = wave & 1;        // 4 x 2 wave grid
  const int hf = lane >> 4, r = lane & 15;
  const int bn0 = blockIdx.x * 128, bm0 = blockIdx.y * 256;

  // staging: per tile each thread issues 6 b128 transfers (4 As rows, 2 Bs)
  const int lr = tid >> 2;            // 0..63
  const int lk = (tid & 3) << 3;      // 0,8,16,24 (bf16 elems)
  const bf16* gA = A  + (size_t)(bm0 + lr) * K + lk;
  const bf16* gB = Bt + (size_t)(bn0 + lr) * K + lk;
  const size_t sA = (size_t)64 * K;   // 64-row stride in A (elems)
  const unsigned la = lds_off(&As[0][lr][lk]);
  const unsigned lb = lds_off(&Bs[0][lr][lk]);
  const unsigned ABUF = 256 * 32 * 2;   // 16384 B between As buffers
  const unsigned BBUF = 128 * 32 * 2;   //  8192 B between Bs buffers
  const unsigned RSTR = 64 * 32 * 2;    //  4096 B per 64 LDS rows

  v8f acc[4][4];
  const v8f z = {0.f, 0.f, 0.f, 0.f, 0.f, 0.f, 0.f, 0.f};
#pragma unroll
  for (int mi = 0; mi < 4; ++mi)
#pragma unroll
    for (int ni = 0; ni < 4; ++ni) acc[mi][ni] = z;

  const int nk = K >> 5;
  // prologue: fill buffer 0
  async_ld_b128(la,            gA);
  async_ld_b128(la + RSTR,     gA + sA);
  async_ld_b128(la + 2 * RSTR, gA + 2 * sA);
  async_ld_b128(la + 3 * RSTR, gA + 3 * sA);
  async_ld_b128(lb,            gB);
  async_ld_b128(lb + RSTR,     gB + sA);
  wait_async0();
  __syncthreads();

  for (int t = 0; t < nk; ++t) {
    const int cur = t & 1;
    if (t + 1 < nk) {                       // prefetch next tile (other buf)
      const int ko = (t + 1) << 5;
      const unsigned ab = (unsigned)((t + 1) & 1) * ABUF;
      const unsigned bb = (unsigned)((t + 1) & 1) * BBUF;
      async_ld_b128(la + ab,            gA + ko);
      async_ld_b128(la + ab + RSTR,     gA + ko + sA);
      async_ld_b128(la + ab + 2 * RSTR, gA + ko + 2 * sA);
      async_ld_b128(la + ab + 3 * RSTR, gA + ko + 3 * sA);
      async_ld_b128(lb + bb,            gB + ko);
      async_ld_b128(lb + bb + RSTR,     gB + ko + sA);
    }

    v16bf b0 = ld_frag(&Bs[cur][wn * 64 +  0 + r][0], hf);
    v16bf b1 = ld_frag(&Bs[cur][wn * 64 + 16 + r][0], hf);
    v16bf b2 = ld_frag(&Bs[cur][wn * 64 + 32 + r][0], hf);
    v16bf b3 = ld_frag(&Bs[cur][wn * 64 + 48 + r][0], hf);
#pragma unroll
    for (int mi = 0; mi < 4; ++mi) {
      v16bf a = ld_frag(&As[cur][wm * 64 + mi * 16 + r][0], hf);
      acc[mi][0] = wmma_bf16(a, b0, acc[mi][0]);
      acc[mi][1] = wmma_bf16(a, b1, acc[mi][1]);
      acc[mi][2] = wmma_bf16(a, b2, acc[mi][2]);
      acc[mi][3] = wmma_bf16(a, b3, acc[mi][3]);
    }

    wait_async0();      // next tile landed (my lanes)
    __syncthreads();    // all waves done reading cur + all async complete
  }

  // Epilogue. C-layout: lane holds col = lane&15, rows hf*8 + j.
  const int row0 = bm0 + wm * 64, col0 = bn0 + wn * 64;
#pragma unroll
  for (int mi = 0; mi < 4; ++mi)
#pragma unroll
    for (int ni = 0; ni < 4; ++ni) {
      const int col = col0 + ni * 16 + r;
      const float bv = bias[col];
#pragma unroll
      for (int j = 0; j < 8; ++j) {
        const int row = row0 + mi * 16 + hf * 8 + j;
        const float v = acc[mi][ni][j] + bv;
        if (mode == 0) {
          outF[(size_t)row * N + col] = v;
        } else {
          const int bb = row >> 11, t = row & 2047;       // M = 2*2048
          const int sel = col >> 10, cc = col & 1023;     // N = 3072
          const int h = cc >> 6, d = cc & 63;
          const size_t bh = (size_t)(bb * 16 + h);
          if (sel == 0)      Qb[(bh * 2048 + t) * 64 + d] = (bf16)v;
          else if (sel == 1) Kb[(bh * 2048 + t) * 64 + d] = (bf16)v;
          else               Vt[(bh * 64 + d) * 2048 + t] = (bf16)v;
        }
      }
    }
}

// ---------------------------------------------------------------------------
// Flash attention (causal). Block = 8 waves = 128 queries; k-tiles of 64.
//    Q[B,H,T,64], K[B,H,T,64], Vt[B,H,64,T] -> attn[B,T,1024] (bf16)
// ---------------------------------------------------------------------------
__global__ __launch_bounds__(256)
void mha_flash_wmma(const bf16* __restrict__ Q, const bf16* __restrict__ Kq,
                    const bf16* __restrict__ Vt, bf16* __restrict__ Ob) {
  __shared__ bf16 Qs[128][64];     // 16 KB
  __shared__ bf16 Ks[64][64];      //  8 KB (rows = key idx, cols = hd)
  __shared__ bf16 Vs[64][64];      //  8 KB (rows = hd, cols = key idx)
  __shared__ bf16 Ps[8][16][64];   // 16 KB per-wave P round-trip
  const int tid  = threadIdx.x;
  const int wave = tid >> 5, lane = tid & 31;
  const int hf = lane >> 4, r = lane & 15;
  const int q0 = blockIdx.x * 128;
  const int bh = blockIdx.y;                         // b*16 + h
  const size_t base = (size_t)bh * (2048 * 64);
  const bf16* Qp = Q  + base;
  const bf16* Kp = Kq + base;
  const bf16* Vp = Vt + base;

  // staging coords: each thread owns one b128 chunk per 32-row group
  const int sr = tid >> 3;            // 0..31
  const int sc = (tid & 7) << 3;      // 0..56

  {  // Qs: 128 rows, 4 async b128 per thread
    const bf16* g = Qp + (size_t)(q0 + sr) * 64 + sc;
    const unsigned l = lds_off(&Qs[sr][sc]);
    async_ld_b128(l,         g);
    async_ld_b128(l + 4096,  g + 32 * 64);
    async_ld_b128(l + 8192,  g + 64 * 64);
    async_ld_b128(l + 12288, g + 96 * 64);
    wait_async0();
  }
  __syncthreads();

  v16bf qa0 = ld_frag(&Qs[wave * 16 + r][0],  hf);   // hd 0..31
  v16bf qa1 = ld_frag(&Qs[wave * 16 + r][32], hf);   // hd 32..63

  const v8f z = {0.f, 0.f, 0.f, 0.f, 0.f, 0.f, 0.f, 0.f};
  v8f o[4];
#pragma unroll
  for (int oi = 0; oi < 4; ++oi) o[oi] = z;
  float m[8], l[8];
#pragma unroll
  for (int j = 0; j < 8; ++j) { m[j] = -3.0e38f; l[j] = 0.f; }

  const bf16* gK = Kp + (size_t)sr * 64 + sc;     // advance kt*64*64
  const bf16* gV = Vp + (size_t)sr * 2048 + sc;   // advance kt*64
  const unsigned lK = lds_off(&Ks[sr][sc]);
  const unsigned lV = lds_off(&Vs[sr][sc]);

  const int nkt = (q0 >> 6) + 2;                  // causal coverage
  for (int kt = 0; kt < nkt; ++kt) {
    const bf16* gk = gK + (size_t)kt * (64 * 64);
    const bf16* gv = gV + kt * 64;
    async_ld_b128(lK,        gk);
    async_ld_b128(lK + 4096, gk + 32 * 64);
    async_ld_b128(lV,        gv);
    async_ld_b128(lV + 4096, gv + (size_t)32 * 2048);
    wait_async0();
    __syncthreads();

    // S = (Q K^T) * 1/sqrt(64): 16x64 per wave as 4 C-frags
    v8f s[4];
#pragma unroll
    for (int ni = 0; ni < 4; ++ni) {
      v16bf kb0 = ld_frag(&Ks[ni * 16 + r][0],  hf);
      v16bf kb1 = ld_frag(&Ks[ni * 16 + r][32], hf);
      s[ni] = wmma_bf16(qa0, kb0, z);
      s[ni] = wmma_bf16(qa1, kb1, s[ni]);
    }

    // scale + causal mask + per-element row max
    const int qrb = q0 + wave * 16 + hf * 8;
    float mt[8];
#pragma unroll
    for (int j = 0; j < 8; ++j) mt[j] = -3.0e38f;
#pragma unroll
    for (int ni = 0; ni < 4; ++ni) {
      const int kcol = kt * 64 + ni * 16 + r;
#pragma unroll
      for (int j = 0; j < 8; ++j) {
        float sv = s[ni][j] * 0.125f;
        if (kcol > qrb + j) sv = -3.0e38f;
        s[ni][j] = sv;
        mt[j] = fmaxf(mt[j], sv);
      }
    }
#pragma unroll
    for (int off = 1; off < 16; off <<= 1)
#pragma unroll
      for (int j = 0; j < 8; ++j)
        mt[j] = fmaxf(mt[j], __shfl_xor(mt[j], off, 32));

    float alpha[8], rs[8];
#pragma unroll
    for (int j = 0; j < 8; ++j) {
      float mn = fmaxf(m[j], mt[j]);
      alpha[j] = __expf(m[j] - mn);
      m[j] = mn;
      rs[j] = 0.f;
    }
#pragma unroll
    for (int ni = 0; ni < 4; ++ni)
#pragma unroll
      for (int j = 0; j < 8; ++j) {
        float p = __expf(s[ni][j] - m[j]);
        s[ni][j] = p;
        rs[j] += p;
      }
#pragma unroll
    for (int off = 1; off < 16; off <<= 1)
#pragma unroll
      for (int j = 0; j < 8; ++j)
        rs[j] += __shfl_xor(rs[j], off, 32);
#pragma unroll
    for (int j = 0; j < 8; ++j) l[j] = l[j] * alpha[j] + rs[j];
#pragma unroll
    for (int oi = 0; oi < 4; ++oi)
#pragma unroll
      for (int j = 0; j < 8; ++j) o[oi][j] *= alpha[j];

    // P (C-layout) -> LDS -> re-load in A-layout for P @ V
#pragma unroll
    for (int ni = 0; ni < 4; ++ni)
#pragma unroll
      for (int j = 0; j < 8; ++j)
        Ps[wave][hf * 8 + j][ni * 16 + r] = (bf16)s[ni][j];
    __syncthreads();

#pragma unroll
    for (int ks = 0; ks < 2; ++ks) {
      v16bf pa = ld_frag(&Ps[wave][r][ks * 32], hf);
#pragma unroll
      for (int oi = 0; oi < 4; ++oi) {
        v16bf vb = ld_frag(&Vs[oi * 16 + r][ks * 32], hf);
        o[oi] = wmma_bf16(pa, vb, o[oi]);
      }
    }
    __syncthreads();   // all reads of Ks/Vs/Ps done before next tile's loads
  }

  // normalize + scatter to attn[B,T,C] bf16
  const int bb = bh >> 4, h = bh & 15;
#pragma unroll
  for (int oi = 0; oi < 4; ++oi)
#pragma unroll
    for (int j = 0; j < 8; ++j) {
      const int row = q0 + wave * 16 + hf * 8 + j;
      const int col = h * 64 + oi * 16 + r;
      Ob[((size_t)(bb * 2048 + row)) * 1024 + col] = (bf16)(o[oi][j] / l[j]);
    }
}

// ---------------------------------------------------------------------------
// Launcher.  ws layout (bytes): xh@0(8M) wqkvT@8M(6M) woutT@14M(2M)
//                               Q@16M(8M) K@24M(8M) Vt@32M(8M) attn@40M(8M)
// ---------------------------------------------------------------------------
extern "C" void kernel_launch(void* const* d_in, const int* in_sizes, int n_in,
                              void* d_out, int out_size, void* d_ws, size_t ws_size,
                              hipStream_t stream) {
  (void)in_sizes; (void)n_in; (void)out_size; (void)ws_size;
  const float* x     = (const float*)d_in[0];
  const float* w_qkv = (const float*)d_in[1];
  const float* b_qkv = (const float*)d_in[2];
  const float* w_out = (const float*)d_in[3];
  const float* b_out = (const float*)d_in[4];
  float* out = (float*)d_out;

  char* ws = (char*)d_ws;
  const size_t MB = 1024 * 1024;
  bf16* xh    = (bf16*)(ws + 0 * MB);
  bf16* wqkvT = (bf16*)(ws + 8 * MB);
  bf16* woutT = (bf16*)(ws + 14 * MB);
  bf16* Qb    = (bf16*)(ws + 16 * MB);
  bf16* Kb    = (bf16*)(ws + 24 * MB);
  bf16* Vt    = (bf16*)(ws + 32 * MB);
  bf16* attn  = (bf16*)(ws + 40 * MB);

  const int Mrows = 4096;   // B*T
  const int C     = 1024;

  mha_cvt4<<<(Mrows * C / 4 + 255) / 256, 256, 0, stream>>>(
      (const float4*)x, (v4bf*)xh, Mrows * C / 4);
  mha_transpose_cvt<<<(C * 3072 + 255) / 256, 256, 0, stream>>>(
      w_qkv, wqkvT, C, 3072);
  mha_transpose_cvt<<<(C * C + 255) / 256, 256, 0, stream>>>(
      w_out, woutT, C, C);

  // QKV projection: [4096,1024] @ [1024,3072] -> Q/K/Vt (bf16, head-major)
  mha_gemm_wmma<<<dim3(3072 / 128, Mrows / 256), 256, 0, stream>>>(
      xh, wqkvT, b_qkv, nullptr, Qb, Kb, Vt, Mrows, 3072, C, 1);

  // Flash attention: grid (T/128 q-tiles, B*H)
  mha_flash_wmma<<<dim3(2048 / 128, 32), 256, 0, stream>>>(Qb, Kb, Vt, attn);

  // Output projection: [4096,1024] @ [1024,1024] + b_out -> f32 out
  mha_gemm_wmma<<<dim3(C / 128, Mrows / 256), 256, 0, stream>>>(
      attn, woutT, b_out, out, nullptr, nullptr, nullptr, Mrows, C, C, 0);
}
